// SE3Transformer_90726889161255
// MI455X (gfx1250) — matrix operand order repaired and checked by reading, not compiled, not run on hardware
//
#include <hip/hip_runtime.h>
#include <hip/hip_bf16.h>
#include <math.h>

#define NEDGE 65536
#define NNODE 8192

typedef __attribute__((ext_vector_type(16))) _Float16 v16h;
typedef __attribute__((ext_vector_type(8)))  float    v8f;

// ---------------------------------------------------------------------------
// WMMA helpers. Fragments are kept in *swizzled* (lane-contiguous) order so
// each lane loads its 32-byte operand slice with wide b128 loads.
//
// A operand (16x32 f16, ISA 7.12.2): lane L holds (m=L&15, half=L>>4);
//   within-lane element idx <-> k = (idx&7) + 16*(idx>>3) + 8*half
// B operand (32x16 f16): lane L holds column n=(L&15); half=L>>4;
//   within-lane element idx <-> k = (idx&7) + 16*(idx>>3) + 8*half
// D/C (16x16 f32): VGPR v -> row M = v + 8*(L>>4), col N = (L&15)
// ---------------------------------------------------------------------------
__device__ inline v8f wmma_f16(v16h a, v16h b, v8f c) {
  return __builtin_amdgcn_wmma_f32_16x16x32_f16(false, a, false, b, (short)0, c, false, false);
}

__device__ inline v16h load_a_sw(const _Float16* hsw) {
  int lane = threadIdx.x & 31;
  return *(const v16h*)(hsw + lane * 16);
}
__device__ inline v16h load_b_sw(const _Float16* base, int nt) {
  int lane = threadIdx.x & 31;
  return *(const v16h*)(base + ((nt << 5) + lane) * 16);
}

// Store one D fragment with BN2 scale/shift + ReLU into the swizzled A layout.
__device__ inline void store_h2_bn(_Float16* hsw, const v8f& d, int n0,
                                   const float* sc, const float* sh) {
  int lane = threadIdx.x & 31; int col = (lane & 15) + n0; int hl = lane >> 4;
  int chalf = (col >> 3) & 1;
  int cidx = (col & 7) + 8 * (col >> 4);
  float s = sc[col], t = sh[col];
#pragma unroll
  for (int v = 0; v < 8; ++v) {
    float x = d[v] * s + t; x = x > 0.f ? x : 0.f;
    int row = v + 8 * hl;
    hsw[((row + 16 * chalf) << 4) + cidx] = (_Float16)x;
  }
}

template <int W3>
__device__ inline void store_r_frag(float* rs, const v8f& d, int n0, const float* b3) {
  int lane = threadIdx.x & 31; int col = (lane & 15) + n0; int hl = lane >> 4;
  float bb = b3[col];
#pragma unroll
  for (int v = 0; v < 8; ++v) rs[(v + 8 * hl) * W3 + col] = d[v] + bb;
}

__device__ inline void accum_cols(const v8f& d, int n0, float* cs, float* cs2) {
  int lane = threadIdx.x & 31; int col = (lane & 15) + n0;
  float s = 0.f, s2 = 0.f;
#pragma unroll
  for (int v = 0; v < 8; ++v) { s += d[v]; s2 += d[v] * d[v]; }
  atomicAdd(cs + col, s);
  atomicAdd(cs2 + col, s2);
}

// ---------------------------------------------------------------------------
// Radial MLP layer 1 (K=5, VALU; BN1 folded into w1f/b1f) -> H1 f16 in LDS
// (written directly in swizzled A-fragment order)
// ---------------------------------------------------------------------------
__device__ inline void radial_h1(const float* __restrict__ ew, const float* __restrict__ rr,
                                 const float* __restrict__ w1f, const float* __restrict__ b1f,
                                 float (*ein)[5], _Float16* hsw, int ebase) {
  int lane = threadIdx.x & 31;
  if (lane < 16) {
    int e = ebase + lane;
    float4 w4 = ((const float4*)ew)[e];
    ein[lane][0] = w4.x; ein[lane][1] = w4.y; ein[lane][2] = w4.z; ein[lane][3] = w4.w;
    ein[lane][4] = rr[e];
  }
  __builtin_amdgcn_wave_barrier();
  int k = lane;                       // this lane computes H1 column k for 16 rows
  int khalf = (k >> 3) & 1;
  int kidx = (k & 7) + 8 * (k >> 4);
  float wa0 = w1f[k], wa1 = w1f[32 + k], wa2 = w1f[64 + k], wa3 = w1f[96 + k], wa4 = w1f[128 + k];
  float bk = b1f[k];
#pragma unroll
  for (int m = 0; m < 16; ++m) {
    float acc = bk + ein[m][0] * wa0 + ein[m][1] * wa1 + ein[m][2] * wa2 +
                ein[m][3] * wa3 + ein[m][4] * wa4;
    hsw[((m + 16 * khalf) << 4) + kidx] = (_Float16)(acc > 0.f ? acc : 0.f);
  }
  __builtin_amdgcn_wave_barrier();
}

// ---------------------------------------------------------------------------
// Small utility kernels
// ---------------------------------------------------------------------------
__global__ void k_fill(float* p, float v, int n) {
  int i = blockIdx.x * blockDim.x + threadIdx.x;
  if (i < n) p[i] = v;
}

// e_in first/second moments: S[0..4]=sum, S[5..29]=sum of outer products
__global__ void k_ein_stats(const float* __restrict__ ew, const float* __restrict__ rr,
                            float* __restrict__ S) {
  __shared__ float sacc[30];
  if (threadIdx.x < 30) sacc[threadIdx.x] = 0.f;
  __syncthreads();
  float p[30];
#pragma unroll
  for (int i = 0; i < 30; ++i) p[i] = 0.f;
  for (int e = blockIdx.x * blockDim.x + threadIdx.x; e < NEDGE; e += gridDim.x * blockDim.x) {
    float4 w4 = ((const float4*)ew)[e];
    float a[5] = {w4.x, w4.y, w4.z, w4.w, rr[e]};
#pragma unroll
    for (int i = 0; i < 5; ++i) {
      p[i] += a[i];
#pragma unroll
      for (int j = 0; j < 5; ++j) p[5 + i * 5 + j] += a[i] * a[j];
    }
  }
#pragma unroll
  for (int i = 0; i < 30; ++i) atomicAdd(&sacc[i], p[i]);
  __syncthreads();
  if (threadIdx.x < 30) atomicAdd(&S[threadIdx.x], sacc[threadIdx.x]);
}

// Fold BN1 into (w1f,b1f) via e_in covariance; convert w2,w3 to f16 in
// pre-swizzled B-fragment order; zero column stats.
__global__ void k_fold_bn1(const float* __restrict__ P, int ob1, int obe1, int og1, int ow1,
                           int ow2, int ow3, int W3, const float* __restrict__ S,
                           float* w1f, float* b1f, _Float16* w2h, _Float16* w3h,
                           float* cs, float* cs2) {
  int j = threadIdx.x;
  if (j < 32) {
    float m[5];
#pragma unroll
    for (int a = 0; a < 5; ++a) m[a] = S[a] * (1.f / NEDGE);
    float mu = P[ob1 + j], var = 0.f;
#pragma unroll
    for (int a = 0; a < 5; ++a) {
      float wa = P[ow1 + a * 32 + j];
      mu += m[a] * wa;
#pragma unroll
      for (int b = 0; b < 5; ++b) {
        float cov = S[5 + a * 5 + b] * (1.f / NEDGE) - m[a] * m[b];
        var += wa * P[ow1 + b * 32 + j] * cov;
      }
    }
    float al = P[og1 + j] * rsqrtf(var + 1e-5f);
#pragma unroll
    for (int a = 0; a < 5; ++a) w1f[a * 32 + j] = P[ow1 + a * 32 + j] * al;
    b1f[j] = (P[ob1 + j] - mu) * al + P[obe1 + j];
    cs[j] = 0.f; cs2[j] = 0.f;
  }
  // B-swizzle: t = (nt*32 + L)*16 + idx ; k = (idx&7) + 16*(idx>>3) + 8*(L>>4) ; n = (L&15)+16*nt
  for (int t = j; t < 1024; t += blockDim.x) {
    int idx = t & 15, L = (t >> 4) & 31, nt = t >> 9;
    int k = (idx & 7) + 16 * (idx >> 3) + 8 * (L >> 4);
    int n = (L & 15) + 16 * nt;
    w2h[t] = (_Float16)P[ow2 + k * 32 + n];
  }
  for (int t = j; t < 32 * W3; t += blockDim.x) {
    int idx = t & 15, L = (t >> 4) & 31, nt = t >> 9;
    int k = (idx & 7) + 16 * (idx >> 3) + 8 * (L >> 4);
    int n = (L & 15) + 16 * nt;
    w3h[t] = (_Float16)P[ow3 + k * W3 + n];
  }
}

__global__ void k_fold_bn2(const float* cs, const float* cs2, const float* __restrict__ P,
                           int og2, int obe2, float* scale2, float* shift2) {
  int j = threadIdx.x; if (j >= 32) return;
  float mean = cs[j] * (1.f / NEDGE);
  float var  = cs2[j] * (1.f / NEDGE) - mean * mean;
  float al = P[og2 + j] * rsqrtf(var + 1e-5f);
  scale2[j] = al;
  shift2[j] = P[obe2 + j] - mean * al;   // b2 cancels (stats collected bias-free)
}

// Stats pass: H1 -> X2 = H1@W2 via WMMA; accumulate column sum / sumsq.
__global__ __launch_bounds__(128) void k_radial_pass1(
    const float* __restrict__ ew, const float* __restrict__ rr,
    const float* __restrict__ w1f, const float* __restrict__ b1f,
    const _Float16* __restrict__ w2h, float* cs, float* cs2) {
  __shared__ alignas(32) float ein[4][16][5];
  __shared__ alignas(32) _Float16 hs[4][512];
  int wave = threadIdx.x >> 5;
  int ebase = (blockIdx.x * 4 + wave) * 16;
  radial_h1(ew, rr, w1f, b1f, ein[wave], hs[wave], ebase);
  v16h a = load_a_sw(hs[wave]);
  v8f z = {};
  v8f c0 = wmma_f16(a, load_b_sw(w2h, 0), z);
  v8f c1 = wmma_f16(a, load_b_sw(w2h, 1), z);
  accum_cols(c0, 0, cs, cs2);
  accum_cols(c1, 16, cs, cs2);
}

// ---------------------------------------------------------------------------
// Fused radial MLP + pairwise equivariant conv.
// MODE 0: V-messages (out[E,CO,OD] +=)    MODE 1: K (logits[E,2] += k.q[dst])
// MODE 2: final conv (out[E,24] +=, OD=1)
// The wave's 16-edge basis block (contiguous in memory) is staged into LDS
// with async global->LDS b128 copies issued at kernel entry; the copy runs
// concurrently with the WMMA radial pipeline and is waited on (ASYNCcnt)
// only right before the per-edge contraction.
// ---------------------------------------------------------------------------
struct PCArgs {
  const float* ew; const float* rr;
  const float* w1f; const float* b1f; const _Float16* w2h;
  const float* scale2; const float* shift2; const _Float16* w3h; const float* b3;
  const float* basis; const float* hsrc; const int* src; const int* dst;
  const float* qbuf; float* out;
};

template <int DI, int DO, int F, int CI, int CO, int MODE, int WPB>
__global__ __launch_bounds__(WPB * 32) void k_pairconv(PCArgs a) {
  constexpr int MD = 2 * DI + 1, OD = 2 * DO + 1, W3 = CO * CI * F, NT = W3 / 16, CPL = CO / 2;
  constexpr int BPE = OD * MD * F;            // basis floats per edge
  constexpr int BBYTES = 16 * BPE * 4;        // basis bytes per wave tile (16B multiple)
  static_assert(W3 % 16 == 0, "W3 tile");
  __shared__ alignas(32) float ein[WPB][16][5];
  __shared__ alignas(32) _Float16 hs[WPB][512];
  __shared__ alignas(32) float rs[WPB][16 * W3];
  __shared__ alignas(32) float basb[WPB][16 * BPE];
  int wave = threadIdx.x >> 5, lane = threadIdx.x & 31;
  int ebase = (blockIdx.x * WPB + wave) * 16;

  // --- kick off async basis tile copy (global -> LDS), tracked by ASYNCcnt ---
  {
    unsigned ldsbase = (unsigned)(unsigned long long)(&basb[wave][0]);
    unsigned gbase = (unsigned)((size_t)ebase * BPE * 4);
    for (int it = lane * 16; it < BBYTES; it += 32 * 16) {
      unsigned laddr = ldsbase + (unsigned)it;
      unsigned voff = gbase + (unsigned)it;
      asm volatile("global_load_async_to_lds_b128 %0, %1, %2"
                   :: "v"(laddr), "v"(voff), "s"(a.basis) : "memory");
    }
  }

  radial_h1(a.ew, a.rr, a.w1f, a.b1f, ein[wave], hs[wave], ebase);
  v16h af = load_a_sw(hs[wave]);
  v8f z = {};
  v8f c0 = wmma_f16(af, load_b_sw(a.w2h, 0), z);
  v8f c1 = wmma_f16(af, load_b_sw(a.w2h, 1), z);
  __builtin_amdgcn_wave_barrier();
  store_h2_bn(hs[wave], c0, 0, a.scale2, a.shift2);
  store_h2_bn(hs[wave], c1, 16, a.scale2, a.shift2);
  __builtin_amdgcn_wave_barrier();
  v16h a2 = load_a_sw(hs[wave]);
#pragma unroll
  for (int nt = 0; nt < NT; ++nt) {
    v8f d = wmma_f16(a2, load_b_sw(a.w3h, nt), z);
    store_r_frag<W3>(rs[wave], d, nt * 16, a.b3);
  }
  __builtin_amdgcn_wave_barrier();

  // --- wait for the async basis tile, then contract per edge ---
  asm volatile("s_wait_asynccnt 0x0" ::: "memory");

  // out[c,o] = sum_{d,f} R[c,d,f] * sum_m basis[o,m,f]*feat[d,m]
  int el = lane >> 1, ch = lane & 1;
  int e = ebase + el;
  int s = a.src[e];
  float feat[CI * MD];
#pragma unroll
  for (int d = 0; d < CI; ++d)
#pragma unroll
    for (int m = 0; m < MD; ++m) feat[d * MD + m] = a.hsrc[(size_t)s * CI * MD + d * MD + m];
  int dd = a.dst[e];
  float lg = 0.f;
  const float* rrow = rs[wave] + el * W3;
  const float* basl = basb[wave] + el * BPE;
#pragma unroll
  for (int o = 0; o < OD; ++o) {
    float bm[MD * F];
#pragma unroll
    for (int j = 0; j < MD * F; ++j) bm[j] = basl[o * MD * F + j];
    float B2[CI * F];
#pragma unroll
    for (int d = 0; d < CI; ++d)
#pragma unroll
      for (int f = 0; f < F; ++f) {
        float acc = 0.f;
#pragma unroll
        for (int m = 0; m < MD; ++m) acc += bm[m * F + f] * feat[d * MD + m];
        B2[d * F + f] = acc;
      }
#pragma unroll
    for (int t = 0; t < CPL; ++t) {
      int c = ch * CPL + t;
      float ov = 0.f;
#pragma unroll
      for (int d = 0; d < CI; ++d)
#pragma unroll
        for (int f = 0; f < F; ++f) ov += rrow[c * CI * F + d * F + f] * B2[d * F + f];
      if constexpr (MODE == 0) a.out[(size_t)e * CO * OD + c * OD + o] += ov;
      else if constexpr (MODE == 2) a.out[(size_t)e * CO + c] += ov;
      else lg += ov * a.qbuf[(size_t)dd * 4 * OD + c * OD + o];
    }
  }
  if constexpr (MODE == 1) a.out[(size_t)e * 2 + ch] += lg;   // head == ch
}

// ---------------------------------------------------------------------------
// Graph / node-wise kernels
// ---------------------------------------------------------------------------
__device__ inline void atomicMaxF(float* addr, float v) {
  if (v >= 0.f) atomicMax((int*)addr, __float_as_int(v));
  else          atomicMin((unsigned int*)addr, __float_as_uint(v));
}

__global__ void k_segmax(const float* logits, const int* dst, float* mx, float scl) {
  int i = blockIdx.x * blockDim.x + threadIdx.x;
  if (i >= NEDGE * 2) return;
  int h = i & 1, e = i >> 1;
  atomicMaxF(&mx[dst[e] * 2 + h], logits[i] * scl);
}
__global__ void k_expsum(const float* logits, const int* dst, const float* mx,
                         float* den, float* att, float scl) {
  int i = blockIdx.x * blockDim.x + threadIdx.x;
  if (i >= NEDGE * 2) return;
  int h = i & 1, e = i >> 1;
  float ex = __expf(logits[i] * scl - mx[dst[e] * 2 + h]);
  att[i] = ex;
  atomicAdd(&den[dst[e] * 2 + h], ex);
}
__global__ void k_attdiv(float* att, const int* dst, const float* den) {
  int i = blockIdx.x * blockDim.x + threadIdx.x;
  if (i >= NEDGE * 2) return;
  int h = i & 1, e = i >> 1;
  att[i] /= den[dst[e] * 2 + h];
}
__global__ void k_agg(const float* msg, const float* att, const int* dst, float* zb,
                      int OD, int total) {
  int i = blockIdx.x * blockDim.x + threadIdx.x;
  if (i >= total) return;
  int o = i % OD, t = i / OD, c = t % 4, e = t / 4;
  atomicAdd(&zb[((size_t)dst[e] * 4 + c) * OD + o], att[e * 2 + (c >> 1)] * msg[i]);
}
__global__ void k_q1x1(const float* __restrict__ W, const float* __restrict__ h,
                       float* q, int MDv, int total) {
  int i = blockIdx.x * blockDim.x + threadIdx.x;
  if (i >= total) return;
  int m = i % MDv, t = i / MDv, c = t % 4, n = t / 4;
  float acc = 0.f;
#pragma unroll
  for (int u = 0; u < 8; ++u) acc += W[c * 8 + u] * h[((size_t)n * 8 + u) * MDv + m];
  q[i] = acc;
}
__global__ void k_proj_res(const float* __restrict__ W, const float* __restrict__ zb,
                           const float* __restrict__ resid, float* y, int MDv, int total) {
  int i = blockIdx.x * blockDim.x + threadIdx.x;
  if (i >= total) return;
  int m = i % MDv, t = i / MDv, c = t % 8, n = t / 8;
  float acc = resid ? resid[i] : 0.f;
#pragma unroll
  for (int u = 0; u < 4; ++u) acc += W[c * 4 + u] * zb[((size_t)n * 4 + u) * MDv + m];
  y[i] = acc;
}
template <int MD>
__global__ void k_gnorm(const float* __restrict__ P, int og, int ob, int ow, int ob2,
                        float* y) {
  int n = blockIdx.x * blockDim.x + threadIdx.x;
  if (n >= NNODE) return;
  float x[8][MD], nf[8];
#pragma unroll
  for (int c = 0; c < 8; ++c) {
    float s = 0.f;
#pragma unroll
    for (int m = 0; m < MD; ++m) { x[c][m] = y[((size_t)n * 8 + c) * MD + m]; s += x[c][m] * x[c][m]; }
    nf[c] = sqrtf(s + 1e-12f);
  }
  float mu = 0.f;
#pragma unroll
  for (int c = 0; c < 8; ++c) mu += nf[c];
  mu *= 0.125f;
  float var = 0.f;
#pragma unroll
  for (int c = 0; c < 8; ++c) { float d0 = nf[c] - mu; var += d0 * d0; }
  var *= 0.125f;
  float rv = rsqrtf(var + 1e-5f);
  float rl[8];
#pragma unroll
  for (int c = 0; c < 8; ++c) {
    float l = (nf[c] - mu) * rv * P[og + c] + P[ob + c];
    rl[c] = l > 0.f ? l : 0.f;
  }
#pragma unroll
  for (int c2 = 0; c2 < 8; ++c2) {
    float sc = P[ob2 + c2];
#pragma unroll
    for (int c = 0; c < 8; ++c) sc += rl[c] * P[ow + c * 8 + c2];
    float inv = sc / nf[c2];
#pragma unroll
    for (int m = 0; m < MD; ++m) y[((size_t)n * 8 + c2) * MD + m] = inv * x[c2][m];
  }
}
__global__ void k_cnt(const int* dst, float* deg) {
  int e = blockIdx.x * blockDim.x + threadIdx.x;
  if (e < NEDGE) atomicAdd(&deg[dst[e]], 1.f);
}
__global__ void k_aggf(const float* msg, const int* dst, float* ho) {
  int i = blockIdx.x * blockDim.x + threadIdx.x;
  if (i >= NEDGE * 24) return;
  int c = i % 24, e = i / 24;
  atomicAdd(&ho[(size_t)dst[e] * 24 + c], msg[i]);
}
__global__ void k_selfint(float* ho, const float* deg, const float* __restrict__ P,
                          int oself, const float* __restrict__ h0) {
  int i = blockIdx.x * blockDim.x + threadIdx.x;
  if (i >= NNODE * 24) return;
  int c = i % 24, n = i / 24;
  float d0 = deg[n]; d0 = d0 < 1.f ? 1.f : d0;
  float acc = ho[i] / d0;
#pragma unroll
  for (int u = 0; u < 8; ++u) acc += P[oself + c * 8 + u] * h0[n * 8 + u];
  ho[i] = acc;
}
__global__ void k_pool_fc(const float* __restrict__ ho, const float* __restrict__ P,
                          int ow1, int ob1, int ow2, int ob2, float* out) {
  __shared__ float pooled[24];
  __shared__ float hid[24];
  int t = threadIdx.x;
  if (t < 24) {
    float s = 0.f;
    for (int n = 0; n < NNODE; ++n) s += ho[(size_t)n * 24 + t];
    pooled[t] = s * (1.f / NNODE);
  }
  __syncthreads();
  if (t < 24) {
    float s = P[ob1 + t];
    for (int c = 0; c < 24; ++c) s += pooled[c] * P[ow1 + c * 24 + t];
    hid[t] = s > 0.f ? s : 0.f;
  }
  __syncthreads();
  if (t < 3) {
    float s = P[ob2 + t];
    for (int o = 0; o < 24; ++o) s += hid[o] * P[ow2 + o * 3 + t];
    out[t] = s;
  }
}

// ---------------------------------------------------------------------------
// Host side
// ---------------------------------------------------------------------------
struct Rad { int b1, b2, b3, be1, be2, g1, g2, w1, w2, w3, W3; };
static Rad takeRad(int& cur, int W3) {
  Rad r; r.W3 = W3;
  r.b1 = cur; cur += 32; r.b2 = cur; cur += 32; r.b3 = cur; cur += W3;
  r.be1 = cur; cur += 32; r.be2 = cur; cur += 32; r.g1 = cur; cur += 32; r.g2 = cur; cur += 32;
  r.w1 = cur; cur += 160; r.w2 = cur; cur += 1024; r.w3 = cur; cur += 32 * W3;
  return r;
}
static inline int frq(int a, int b) { int m = a < b ? a : b; return 2 * m + 1; }

template <int MODE>
static void launch_pc(int di, int do_, const PCArgs& a, hipStream_t s) {
  dim3 g4(NEDGE / 64), b4(128);
  dim3 g2(NEDGE / 32), b2(64);
  switch (di * 3 + do_) {
    case 0: k_pairconv<0, 0, 1, 8, 4, MODE, 4><<<g4, b4, 0, s>>>(a); break;
    case 1: k_pairconv<0, 1, 1, 8, 4, MODE, 4><<<g4, b4, 0, s>>>(a); break;
    case 2: k_pairconv<0, 2, 1, 8, 4, MODE, 4><<<g4, b4, 0, s>>>(a); break;
    case 3: k_pairconv<1, 0, 1, 8, 4, MODE, 4><<<g4, b4, 0, s>>>(a); break;
    case 4: k_pairconv<1, 1, 3, 8, 4, MODE, 4><<<g4, b4, 0, s>>>(a); break;
    case 5: k_pairconv<1, 2, 3, 8, 4, MODE, 4><<<g4, b4, 0, s>>>(a); break;
    case 6: k_pairconv<2, 0, 1, 8, 4, MODE, 4><<<g4, b4, 0, s>>>(a); break;
    case 7: k_pairconv<2, 1, 3, 8, 4, MODE, 4><<<g4, b4, 0, s>>>(a); break;
    case 8: k_pairconv<2, 2, 5, 8, 4, MODE, 2><<<g2, b2, 0, s>>>(a); break;  // big LDS tile
  }
}
static void launch_pc_final(int di, const PCArgs& a, hipStream_t s) {
  dim3 g(NEDGE / 64), b(128);
  switch (di) {
    case 0: k_pairconv<0, 0, 1, 8, 24, 2, 4><<<g, b, 0, s>>>(a); break;
    case 1: k_pairconv<1, 0, 1, 8, 24, 2, 4><<<g, b, 0, s>>>(a); break;
    case 2: k_pairconv<2, 0, 1, 8, 24, 2, 4><<<g, b, 0, s>>>(a); break;
  }
}
static void launch_gnorm(int MD, const float* P, int og, int ob, int ow, int ob2,
                         float* y, hipStream_t s) {
  dim3 g((NNODE + 127) / 128), b(128);
  switch (MD) {
    case 1: k_gnorm<1><<<g, b, 0, s>>>(P, og, ob, ow, ob2, y); break;
    case 3: k_gnorm<3><<<g, b, 0, s>>>(P, og, ob, ow, ob2, y); break;
    case 5: k_gnorm<5><<<g, b, 0, s>>>(P, og, ob, ow, ob2, y); break;
  }
}

extern "C" void kernel_launch(void* const* d_in, const int* in_sizes, int n_in,
                              void* d_out, int out_size, void* d_ws, size_t ws_size,
                              hipStream_t stream) {
  (void)in_sizes; (void)n_in; (void)out_size; (void)ws_size;
  const float* node_feat = (const float*)d_in[0];
  const float* ew = (const float*)d_in[1];
  const float* rr = (const float*)d_in[2];
  const float* basis[3][3];
  for (int di = 0; di < 3; ++di)
    for (int d2 = 0; d2 < 3; ++d2) basis[di][d2] = (const float*)d_in[3 + di * 3 + d2];
  const float* P = (const float*)d_in[12];
  const int* srci = (const int*)d_in[13];
  const int* dsti = (const int*)d_in[14];
  float* out = (float*)d_out;

  // ---- params blob offsets (JAX pytree flatten: dict keys sorted) ----
  int cur = 0;
  int fc_b1 = cur; cur += 24; int fc_b2 = cur; cur += 3;
  int fc_w1 = cur; cur += 576; int fc_w2 = cur; cur += 72;
  Rad fin[3];
  for (int di = 0; di < 3; ++di) fin[di] = takeRad(cur, 192);
  int o_self = cur; cur += 192;
  // layer 0: keys k < norm < proj < q < v
  Rad L0k = takeRad(cur, 32);
  int L0n_b[3], L0n_b2[3], L0n_g[3], L0n_w[3];
  for (int d = 0; d < 3; ++d) { L0n_b[d] = cur; cur += 8; L0n_b2[d] = cur; cur += 8;
                                L0n_g[d] = cur; cur += 8; L0n_w[d] = cur; cur += 64; }
  int L0proj[3]; for (int d = 0; d < 3; ++d) { L0proj[d] = cur; cur += 32; }
  int L0q = cur; cur += 32;
  Rad L0v[3]; for (int d = 0; d < 3; ++d) L0v[d] = takeRad(cur, 32);
  // layer 1
  Rad L1k[3][3];
  for (int di = 0; di < 3; ++di) for (int d2 = 0; d2 < 3; ++d2)
    L1k[di][d2] = takeRad(cur, 32 * frq(di, d2));
  int L1n_b[3], L1n_b2[3], L1n_g[3], L1n_w[3];
  for (int d = 0; d < 3; ++d) { L1n_b[d] = cur; cur += 8; L1n_b2[d] = cur; cur += 8;
                                L1n_g[d] = cur; cur += 8; L1n_w[d] = cur; cur += 64; }
  int L1proj[3]; for (int d = 0; d < 3; ++d) { L1proj[d] = cur; cur += 32; }
  int L1q[3];    for (int d = 0; d < 3; ++d) { L1q[d] = cur; cur += 32; }
  Rad L1v[3][3];
  for (int di = 0; di < 3; ++di) for (int d2 = 0; d2 < 3; ++d2)
    L1v[di][d2] = takeRad(cur, 32 * frq(di, d2));

  // ---- workspace carve (all chunks are multiples of 8 floats = 32B) ----
  float* W = (float*)d_ws; size_t wc = 0;
  auto alloc = [&](size_t n) { float* p = W + wc; wc += n; return p; };
  float* S = alloc(32);
  float* w1f = alloc(160); float* b1f = alloc(32);
  float* scale2 = alloc(32); float* shift2 = alloc(32);
  float* csum = alloc(32); float* csq = alloc(32);
  _Float16* w2h = (_Float16*)alloc(512);
  _Float16* w3h = (_Float16*)alloc(3072);
  float* qb[3] = {alloc(NNODE * 4), alloc(NNODE * 12), alloc(NNODE * 20)};
  float* msgv[3] = {alloc(NEDGE * 4), alloc(NEDGE * 12), alloc(NEDGE * 20)};
  float* logits = alloc(NEDGE * 2); float* att = alloc(NEDGE * 2);
  float* mx = alloc(NNODE * 2); float* den = alloc(NNODE * 2);
  float* zb[3] = {alloc(NNODE * 4), alloc(NNODE * 12), alloc(NNODE * 20)};
  float* hA[3] = {alloc(NNODE * 8), alloc(NNODE * 24), alloc(NNODE * 40)};
  float* hB[3] = {alloc(NNODE * 8), alloc(NNODE * 24), alloc(NNODE * 40)};
  float* msgf = alloc(NEDGE * 24);
  float* deg = alloc(NNODE);
  float* ho = alloc(NNODE * 24);

  auto fill = [&](float* p, float v, size_t n) {
    k_fill<<<dim3((unsigned)((n + 255) / 256)), dim3(256), 0, stream>>>(p, v, (int)n);
  };
  auto prep = [&](const Rad& rp) {
    k_fold_bn1<<<1, 256, 0, stream>>>(P, rp.b1, rp.be1, rp.g1, rp.w1, rp.w2, rp.w3, rp.W3,
                                      S, w1f, b1f, w2h, w3h, csum, csq);
    k_radial_pass1<<<NEDGE / 64, 128, 0, stream>>>(ew, rr, w1f, b1f, w2h, csum, csq);
    k_fold_bn2<<<1, 32, 0, stream>>>(csum, csq, P, rp.g2, rp.be2, scale2, shift2);
  };
  auto pcargs = [&](const float* bas, const float* hsrc, const float* q, float* o2,
                    const Rad& rp) {
    PCArgs a; a.ew = ew; a.rr = rr; a.w1f = w1f; a.b1f = b1f; a.w2h = w2h;
    a.scale2 = scale2; a.shift2 = shift2; a.w3h = w3h; a.b3 = P + rp.b3;
    a.basis = bas; a.hsrc = hsrc; a.src = srci; a.dst = dsti; a.qbuf = q; a.out = o2;
    return a;
  };
  auto softmax = [&](float scl) {
    fill(mx, -3e38f, NNODE * 2); fill(den, 0.f, NNODE * 2);
    k_segmax<<<512, 256, 0, stream>>>(logits, dsti, mx, scl);
    k_expsum<<<512, 256, 0, stream>>>(logits, dsti, mx, den, att, scl);
    k_attdiv<<<512, 256, 0, stream>>>(att, dsti, den);
  };

  // ---- e_in statistics (shared by every radial BN1 fold) ----
  fill(S, 0.f, 32);
  k_ein_stats<<<64, 256, 0, stream>>>(ew, rr, S);

  // ================= layer 0 : in_degs={0}, mid={0} =================
  k_q1x1<<<(NNODE * 4 + 255) / 256, 256, 0, stream>>>(P + L0q, node_feat, qb[0], 1, NNODE * 4);
  fill(logits, 0.f, NEDGE * 2);
  prep(L0k);
  launch_pc<1>(0, 0, pcargs(basis[0][0], node_feat, qb[0], logits, L0k), stream);
  softmax(0.5f);  // 1/sqrt(nkey), nkey = 4
  for (int d2 = 0; d2 < 3; ++d2) {
    int OD = 2 * d2 + 1;
    fill(msgv[d2], 0.f, (size_t)NEDGE * 4 * OD);
    prep(L0v[d2]);
    launch_pc<0>(0, d2, pcargs(basis[0][d2], node_feat, nullptr, msgv[d2], L0v[d2]), stream);
    fill(zb[d2], 0.f, (size_t)NNODE * 4 * OD);
    int tot = NEDGE * 4 * OD;
    k_agg<<<(tot + 255) / 256, 256, 0, stream>>>(msgv[d2], att, dsti, zb[d2], OD, tot);
    int ty = NNODE * 8 * OD;
    k_proj_res<<<(ty + 255) / 256, 256, 0, stream>>>(P + L0proj[d2], zb[d2],
                                                     d2 == 0 ? node_feat : nullptr,
                                                     hA[d2], OD, ty);
    launch_gnorm(OD, P, L0n_g[d2], L0n_b[d2], L0n_w[d2], L0n_b2[d2], hA[d2], stream);
  }

  // ================= layer 1 : in_degs={0,1,2}, mid={0,1,2} =================
  for (int d = 0; d < 3; ++d) {
    int OD = 2 * d + 1, tot = NNODE * 4 * OD;
    k_q1x1<<<(tot + 255) / 256, 256, 0, stream>>>(P + L1q[d], hA[d], qb[d], OD, tot);
  }
  fill(logits, 0.f, NEDGE * 2);
  for (int di = 0; di < 3; ++di)
    for (int d2 = 0; d2 < 3; ++d2) {
      prep(L1k[di][d2]);
      launch_pc<1>(di, d2, pcargs(basis[di][d2], hA[di], qb[d2], logits, L1k[di][d2]), stream);
    }
  softmax(1.f / 6.f);  // nkey = 36
  for (int d2 = 0; d2 < 3; ++d2)
    fill(msgv[d2], 0.f, (size_t)NEDGE * 4 * (2 * d2 + 1));
  for (int di = 0; di < 3; ++di)
    for (int d2 = 0; d2 < 3; ++d2) {
      prep(L1v[di][d2]);
      launch_pc<0>(di, d2, pcargs(basis[di][d2], hA[di], nullptr, msgv[d2], L1v[di][d2]), stream);
    }
  for (int d2 = 0; d2 < 3; ++d2) {
    int OD = 2 * d2 + 1;
    fill(zb[d2], 0.f, (size_t)NNODE * 4 * OD);
    int tot = NEDGE * 4 * OD;
    k_agg<<<(tot + 255) / 256, 256, 0, stream>>>(msgv[d2], att, dsti, zb[d2], OD, tot);
    int ty = NNODE * 8 * OD;
    k_proj_res<<<(ty + 255) / 256, 256, 0, stream>>>(P + L1proj[d2], zb[d2], hA[d2],
                                                     hB[d2], OD, ty);
    launch_gnorm(OD, P, L1n_g[d2], L1n_b[d2], L1n_w[d2], L1n_b2[d2], hB[d2], stream);
  }

  // ================= final GConv -> degree 0, mean agg + self =================
  fill(msgf, 0.f, (size_t)NEDGE * 24);
  for (int di = 0; di < 3; ++di) {
    prep(fin[di]);
    launch_pc_final(di, pcargs(basis[di][0], hB[di], nullptr, msgf, fin[di]), stream);
  }
  fill(deg, 0.f, NNODE);
  k_cnt<<<NEDGE / 256, 256, 0, stream>>>(dsti, deg);
  fill(ho, 0.f, (size_t)NNODE * 24);
  k_aggf<<<(NEDGE * 24 + 255) / 256, 256, 0, stream>>>(msgf, dsti, ho);
  k_selfint<<<(NNODE * 24 + 255) / 256, 256, 0, stream>>>(ho, deg, P, o_self, hB[0]);
  k_pool_fc<<<1, 32, 0, stream>>>(ho, P, fc_w1, fc_b1, fc_w2, fc_b2, out);
}
